// Token_QK_Attention_64407329571105
// MI455X (gfx1250) — matrix operand order repaired and checked by reading, not compiled, not run on hardware
//
#include <hip/hip_runtime.h>
#include <hip/hip_bf16.h>

// ---------------------------------------------------------------------------
// Token QK Attention (spiking) for MI455X / gfx1250, bf16 WMMA path.
//   T=4, B=32, C=256, N=1024 (H=W=32), heads=8, dh=32, tau=2, eps=1e-5
// WG owns (b, 32-col n-chunk), loops t; LIF membrane state lives in the WMMA
// f32 accumulator VGPR layout across time steps. Tile sized so acc+membrane
// = 64 VGPRs and the (unroll-capped) k-loop cannot spill.
// ---------------------------------------------------------------------------

#define TT 4
#define BB 32
#define CC 256
#define NN 1024
#define HEADS 8
#define DHH 32
#define NCHUNK 32
#define PITCH 264           // 256 + 8 bf16 pad: 528B rows, 16B-aligned, bank-spread
#define EPSBN 1e-5f

typedef __bf16 bf16_t;
typedef __attribute__((ext_vector_type(16))) __bf16 v16bf;
typedef __attribute__((ext_vector_type(8)))  __bf16 v8bf;
typedef __attribute__((ext_vector_type(8)))  float  v8f;

static __device__ __forceinline__ v16bf cat8(v8bf lo, v8bf hi) {
  v16bf r;
#pragma unroll
  for (int i = 0; i < 8; ++i) { r[i] = lo[i]; r[i + 8] = hi[i]; }
  return r;
}

// ------------------------- weight pack: f32 -> bf16 -------------------------
__global__ __launch_bounds__(256) void prep_weights(
    const float* __restrict__ qw, const float* __restrict__ kw,
    const float* __restrict__ pw,
    bf16_t* __restrict__ dq, bf16_t* __restrict__ dk, bf16_t* __restrict__ dp) {
  int i = blockIdx.x * 256 + threadIdx.x;       // 3 * 65536 threads
  int sel = i >> 16;
  int r = i & 65535;
  const float* s = (sel == 0) ? qw : (sel == 1) ? kw : pw;
  bf16_t* d = (sel == 0) ? dq : (sel == 1) ? dk : dp;
  d[r] = (bf16_t)s[r];
}

// ---------------- fold BN (+ projection bias) into scale/shift --------------
__global__ __launch_bounds__(256) void prep_scales(
    const float* qg, const float* qb, const float* qm, const float* qv,
    const float* kg, const float* kb, const float* km, const float* kv,
    const float* pg, const float* pb, const float* pm, const float* pv,
    const float* pbias,
    float* qs, float* qsh, float* ks, float* ksh, float* ps, float* psh) {
  int c = threadIdx.x;
  float s;
  s = qg[c] * rsqrtf(qv[c] + EPSBN); qs[c] = s; qsh[c] = qb[c] - qm[c] * s;
  s = kg[c] * rsqrtf(kv[c] + EPSBN); ks[c] = s; ksh[c] = kb[c] - km[c] * s;
  s = pg[c] * rsqrtf(pv[c] + EPSBN); ps[c] = s;
  psh[c] = pb[c] + (pbias[c] - pm[c]) * s;      // BN(Wx + b): fold bias
}

// --------------- Q & K branches: GEMM + BN + LIF, fused over T --------------
// grid (32, 32): blockIdx.x = n-chunk, blockIdx.y = b. 512 threads = 16 waves,
// waves 0-7 compute Q (rows wv*32..+31), waves 8-15 compute K; one shared
// LDS-staged x tile per WG.
__global__ __launch_bounds__(512) void qk_gemm_lif(
    const float* __restrict__ x,
    const bf16_t* __restrict__ wq, const bf16_t* __restrict__ wk,
    const float* __restrict__ qs, const float* __restrict__ qsh,
    const float* __restrict__ ks, const float* __restrict__ ksh,
    bf16_t* __restrict__ sq, bf16_t* __restrict__ sk) {
  __shared__ alignas(16) bf16_t xt[NCHUNK * PITCH];   // x-tile, [n][c], bf16

  const int tid = threadIdx.x;
  const int n0 = blockIdx.x * NCHUNK;
  const int b = blockIdx.y;
  const int wv = tid >> 5;
  const int lane = tid & 31;
  const int branch = wv >> 3;                // 0 = Q, 1 = K
  const int m0 = (wv & 7) * 32;
  const int laneM = lane & 15;
  const int hi = lane >> 4;

  const bf16_t* W = branch ? wk : wq;
  const float* scl = branch ? ks : qs;
  const float* sh = branch ? ksh : qsh;
  bf16_t* sdst = branch ? sk : sq;

  v8f v[2][2];                               // LIF membrane, accumulator layout
#pragma unroll
  for (int tm = 0; tm < 2; ++tm)
#pragma unroll
    for (int tn = 0; tn < 2; ++tn) { v8f z = {}; v[tm][tn] = z; }

  for (int t = 0; t < TT; ++t) {
    // ---- stage x[t,b,:,n0:n0+32] into LDS, f32->bf16, transposed to [n][c]
    const float* xsrc = x + ((size_t)(t * BB + b) * CC) * NN + n0;
    for (int i = tid; i < CC * NCHUNK; i += 512) {
      int c = i >> 5;
      int nn = i & (NCHUNK - 1);
      xt[nn * PITCH + c] = (bf16_t)xsrc[c * NN + nn];
    }
    if (t + 1 < TT) {                        // warm next tile into cache
      const float* nx = xsrc + (size_t)BB * CC * NN;
      __builtin_prefetch(nx + (tid << 4), 0, 1);
    }
    __syncthreads();

    v8f acc[2][2];
#pragma unroll
    for (int tm = 0; tm < 2; ++tm)
#pragma unroll
      for (int tn = 0; tn < 2; ++tn) { v8f z = {}; acc[tm][tn] = z; }

#pragma unroll 2
    for (int k8 = 0; k8 < 8; ++k8) {
      const int k0 = k8 * 32;
      // A fragments (weights, 16x32 bf16 lane layout):
      //   lanes 0-15: K k0..k0+7 | k0+16..k0+23 ; lanes 16-31: +8
      v16bf a[2];
#pragma unroll
      for (int tm = 0; tm < 2; ++tm) {
        const bf16_t* wpn = W + (m0 + 16 * tm + laneM) * CC + k0 + 8 * hi;
        a[tm] = cat8(*(const v8bf*)wpn, *(const v8bf*)(wpn + 16));
      }
      // B fragments (x-tile, 32x16 bf16): lane = column n, K same split as A
      v16bf bf[2];
#pragma unroll
      for (int tn = 0; tn < 2; ++tn) {
        const bf16_t* bp = xt + (16 * tn + laneM) * PITCH + k0 + 8 * hi;
        bf[tn] = cat8(*(const v8bf*)bp, *(const v8bf*)(bp + 16));
      }
#pragma unroll
      for (int tm = 0; tm < 2; ++tm)
#pragma unroll
        for (int tn = 0; tn < 2; ++tn)
          acc[tm][tn] = __builtin_amdgcn_wmma_f32_16x16x32_bf16(
              false, a[tm], false, bf[tn], (short)0, acc[tm][tn], false, false);
    }
    __syncthreads();

    // ---- epilogue: BN -> LIF (hard reset, tau=2, vth=1) -> bf16 spike store
#pragma unroll
    for (int tm = 0; tm < 2; ++tm)
#pragma unroll
      for (int tn = 0; tn < 2; ++tn)
#pragma unroll
        for (int j = 0; j < 8; ++j) {
          int o = m0 + 16 * tm + j + 8 * hi;
          float y = acc[tm][tn][j] * scl[o] + sh[o];
          float vv = v[tm][tn][j];
          vv += 0.5f * (y - vv);
          float sp = (vv >= 1.0f) ? 1.0f : 0.0f;
          v[tm][tn][j] = vv * (1.0f - sp);
          int n = n0 + 16 * tn + laneM;
          sdst[((t * BB + b) * CC + o) * NN + n] = (bf16_t)sp;
        }
  }
}

// ---- per-head channel sum -> LIF(0.5) -> mask K spikes (in-place over s_q) --
__global__ __launch_bounds__(256) void attn_mul(const bf16_t* sq,
                                                const bf16_t* __restrict__ sk,
                                                bf16_t* xone) {
  int idx = blockIdx.x * 256 + threadIdx.x;   // B*HEADS*N = 262144
  int n = idx & (NN - 1);
  int h = (idx >> 10) & (HEADS - 1);
  int b = idx >> 13;
  float v = 0.f;
  for (int t = 0; t < TT; ++t) {
    int base = ((t * BB + b) * CC + h * DHH) * NN + n;
    float s = 0.f;
#pragma unroll
    for (int d = 0; d < DHH; ++d) s += (float)sq[base + d * NN];
    v += 0.5f * (s - v);
    float att = (v >= 0.5f) ? 1.0f : 0.0f;
    v *= (1.0f - att);
#pragma unroll
    for (int d = 0; d < DHH; ++d) {
      bf16_t kk = sk[base + d * NN];
      xone[base + d * NN] = (att > 0.f) ? kk : (bf16_t)0.0f;
    }
  }
}

// ------------- projection: GEMM + bias/BN + LIF -> f32 spikes out -----------
// grid (32, 32), 256 threads = 8 waves, wave wv covers rows wv*32..+31.
__global__ __launch_bounds__(256) void proj_gemm_lif(
    const bf16_t* __restrict__ xone, const bf16_t* __restrict__ wp,
    const float* __restrict__ ps, const float* __restrict__ psh,
    float* __restrict__ out) {
  __shared__ alignas(16) bf16_t xt[NCHUNK * PITCH];

  const int tid = threadIdx.x;
  const int n0 = blockIdx.x * NCHUNK;
  const int b = blockIdx.y;
  const int wv = tid >> 5;
  const int lane = tid & 31;
  const int m0 = wv * 32;
  const int laneM = lane & 15;
  const int hi = lane >> 4;

  v8f v[2][2];
#pragma unroll
  for (int tm = 0; tm < 2; ++tm)
#pragma unroll
    for (int tn = 0; tn < 2; ++tn) { v8f z = {}; v[tm][tn] = z; }

  for (int t = 0; t < TT; ++t) {
    const bf16_t* xsrc = xone + ((size_t)(t * BB + b) * CC) * NN + n0;
    for (int i = tid; i < CC * NCHUNK; i += 256) {
      int c = i >> 5;
      int nn = i & (NCHUNK - 1);
      xt[nn * PITCH + c] = xsrc[c * NN + nn];
    }
    __syncthreads();

    v8f acc[2][2];
#pragma unroll
    for (int tm = 0; tm < 2; ++tm)
#pragma unroll
      for (int tn = 0; tn < 2; ++tn) { v8f z = {}; acc[tm][tn] = z; }

#pragma unroll 2
    for (int k8 = 0; k8 < 8; ++k8) {
      const int k0 = k8 * 32;
      v16bf a[2];
#pragma unroll
      for (int tm = 0; tm < 2; ++tm) {
        const bf16_t* wpn = wp + (m0 + 16 * tm + laneM) * CC + k0 + 8 * hi;
        a[tm] = cat8(*(const v8bf*)wpn, *(const v8bf*)(wpn + 16));
      }
      v16bf bf[2];
#pragma unroll
      for (int tn = 0; tn < 2; ++tn) {
        const bf16_t* bp = xt + (16 * tn + laneM) * PITCH + k0 + 8 * hi;
        bf[tn] = cat8(*(const v8bf*)bp, *(const v8bf*)(bp + 16));
      }
#pragma unroll
      for (int tm = 0; tm < 2; ++tm)
#pragma unroll
        for (int tn = 0; tn < 2; ++tn)
          acc[tm][tn] = __builtin_amdgcn_wmma_f32_16x16x32_bf16(
              false, a[tm], false, bf[tn], (short)0, acc[tm][tn], false, false);
    }
    __syncthreads();

#pragma unroll
    for (int tm = 0; tm < 2; ++tm)
#pragma unroll
      for (int tn = 0; tn < 2; ++tn)
#pragma unroll
        for (int j = 0; j < 8; ++j) {
          int o = m0 + 16 * tm + j + 8 * hi;
          float y = acc[tm][tn][j] * ps[o] + psh[o];
          float vv = v[tm][tn][j];
          vv += 0.5f * (y - vv);
          float sp = (vv >= 1.0f) ? 1.0f : 0.0f;
          v[tm][tn][j] = vv * (1.0f - sp);
          int n = n0 + 16 * tn + laneM;
          out[((t * BB + b) * CC + o) * NN + n] = sp;
        }
  }
}

// ---------------------------------------------------------------------------
extern "C" void kernel_launch(void* const* d_in, const int* in_sizes, int n_in,
                              void* d_out, int out_size, void* d_ws,
                              size_t ws_size, hipStream_t stream) {
  (void)in_sizes; (void)n_in; (void)out_size; (void)ws_size;
  const float* x = (const float*)d_in[0];
  const float* q_w = (const float*)d_in[1];
  const float* q_gamma = (const float*)d_in[2];
  const float* q_beta = (const float*)d_in[3];
  const float* q_mean = (const float*)d_in[4];
  const float* q_var = (const float*)d_in[5];
  const float* k_w = (const float*)d_in[6];
  const float* k_gamma = (const float*)d_in[7];
  const float* k_beta = (const float*)d_in[8];
  const float* k_mean = (const float*)d_in[9];
  const float* k_var = (const float*)d_in[10];
  const float* p_w = (const float*)d_in[11];
  const float* p_b = (const float*)d_in[12];
  const float* p_gamma = (const float*)d_in[13];
  const float* p_beta = (const float*)d_in[14];
  const float* p_mean = (const float*)d_in[15];
  const float* p_var = (const float*)d_in[16];

  char* ws = (char*)d_ws;
  bf16_t* wq = (bf16_t*)(ws + 0);            // 128 KiB
  bf16_t* wk = (bf16_t*)(ws + 131072);       // 128 KiB
  bf16_t* wp = (bf16_t*)(ws + 262144);       // 128 KiB
  float* qs = (float*)(ws + 393216);
  float* qsh = qs + 256;
  float* ks = qs + 512;
  float* ksh = qs + 768;
  float* ps = qs + 1024;
  float* psh = qs + 1280;
  const size_t SPIKES = (size_t)TT * BB * CC * NN;     // 33.5M elems
  bf16_t* sq = (bf16_t*)(ws + (1u << 20));             // 64 MiB
  bf16_t* sk = sq + SPIKES;                            // 64 MiB

  prep_weights<<<768, 256, 0, stream>>>(q_w, k_w, p_w, wq, wk, wp);
  prep_scales<<<1, 256, 0, stream>>>(q_gamma, q_beta, q_mean, q_var,
                                     k_gamma, k_beta, k_mean, k_var,
                                     p_gamma, p_beta, p_mean, p_var, p_b,
                                     qs, qsh, ks, ksh, ps, psh);
  qk_gemm_lif<<<dim3(32, 32), 512, 0, stream>>>(x, wq, wk, qs, qsh, ks, ksh,
                                                sq, sk);
  // x_one written in place over s_q (each element read+written by same thread)
  attn_mul<<<1024, 256, 0, stream>>>(sq, sk, sq);
  proj_gemm_lif<<<dim3(32, 32), 256, 0, stream>>>(sq, wp, ps, psh,
                                                  (float*)d_out);
}
